// S2VSingle_78005196030026
// MI455X (gfx1250) — matrix operand order repaired and checked by reading, not compiled, not run on hardware
//
#include <hip/hip_runtime.h>

#define DDIM 128
#define PACK_PER 16384  // 4 ktiles * 8 ntiles * 32 lanes * 16 elems per 128x128 matrix

typedef __attribute__((ext_vector_type(16))) __bf16 v16bf;
typedef __attribute__((ext_vector_type(8)))  float v8f;
typedef __attribute__((ext_vector_type(4)))  unsigned int v4u;
typedef __attribute__((ext_vector_type(4)))  unsigned int u32x4;
typedef __attribute__((ext_vector_type(8)))  int i32x8;
typedef __attribute__((ext_vector_type(4)))  int i32x4;

union FragU { v16bf v; v4u u[2]; };

__device__ __forceinline__ unsigned short f2bf(float x) {
  unsigned int u = __float_as_uint(x);
  u += 0x7FFFu + ((u >> 16) & 1u);          // round-to-nearest-even
  return (unsigned short)(u >> 16);
}

__device__ __forceinline__ void atomicMaxF(float* addr, float val) {
  int* ia = (int*)addr;
  int old = __float_as_int(*addr);
  while (__int_as_float(old) < val) {
    int assumed = old;
    old = atomicCAS(ia, assumed, __float_as_int(val));
    if (old == assumed) break;
  }
}

// ---------------- utility fills / conversions ----------------

__global__ void fill_kernel(float* __restrict__ p, float v, long long n) {
  long long i = (long long)blockIdx.x * blockDim.x + threadIdx.x;
  if (i < n) p[i] = v;
}

__global__ void convert_bf16x4_kernel(const float* __restrict__ x,
                                      unsigned short* __restrict__ y, long long n4) {
  long long q = (long long)blockIdx.x * blockDim.x + threadIdx.x;
  if (q >= n4) return;
  long long i = q * 4;
  float4 v = *(const float4*)(x + i);
  uint2 pk;
  pk.x = (unsigned)f2bf(v.x) | ((unsigned)f2bf(v.y) << 16);
  pk.y = (unsigned)f2bf(v.z) | ((unsigned)f2bf(v.w) << 16);
  *(uint2*)(y + i) = pk;
}

// ---------------- CSR build: counts + per-node edge-feature sums ----------------

__global__ void csr_count_kernel(const int* __restrict__ eto, const float* __restrict__ ef,
                                 int* __restrict__ deg, float* __restrict__ efsum, int E) {
  int e = blockIdx.x * blockDim.x + threadIdx.x;
  if (e >= E) return;
  int to = eto[e];
  atomicAdd(&deg[to], 1);
  atomicAdd(&efsum[(size_t)to * 4 + 0], ef[(size_t)e * 3 + 0]);
  atomicAdd(&efsum[(size_t)to * 4 + 1], ef[(size_t)e * 3 + 1]);
  atomicAdd(&efsum[(size_t)to * 4 + 2], ef[(size_t)e * 3 + 2]);
}

__global__ __launch_bounds__(256) void scan1_kernel(const int* __restrict__ deg,
                                                    int* __restrict__ part,
                                                    int* __restrict__ blockSums, int n) {
  __shared__ int s[256];
  int i = blockIdx.x * 256 + threadIdx.x;
  int v = (i < n) ? deg[i] : 0;
  s[threadIdx.x] = v;
  __syncthreads();
  for (int o = 1; o < 256; o <<= 1) {
    int t = (threadIdx.x >= o) ? s[threadIdx.x - o] : 0;
    __syncthreads();
    s[threadIdx.x] += t;
    __syncthreads();
  }
  if (i < n) part[i] = s[threadIdx.x] - v;     // exclusive
  if (threadIdx.x == 255) blockSums[blockIdx.x] = s[255];
}

__global__ __launch_bounds__(512) void scan2_kernel(int* __restrict__ blockSums, int nb) {
  __shared__ int s[512];
  int v = (threadIdx.x < nb) ? blockSums[threadIdx.x] : 0;
  s[threadIdx.x] = v;
  __syncthreads();
  for (int o = 1; o < 512; o <<= 1) {
    int t = (threadIdx.x >= o) ? s[threadIdx.x - o] : 0;
    __syncthreads();
    s[threadIdx.x] += t;
    __syncthreads();
  }
  if (threadIdx.x < nb) blockSums[threadIdx.x] = s[threadIdx.x] - v;  // exclusive
}

__global__ __launch_bounds__(256) void scan3_kernel(int* __restrict__ rowStart,
                                                    const int* __restrict__ blockOffs,
                                                    int* __restrict__ cursor, int n) {
  int i = blockIdx.x * 256 + threadIdx.x;
  if (i >= n) return;
  int v = rowStart[i] + blockOffs[blockIdx.x];
  rowStart[i] = v;
  cursor[i] = v;
}

__global__ void csr_fill_kernel(const int* __restrict__ efrom, const int* __restrict__ eto,
                                int* __restrict__ cursor, int* __restrict__ src_sorted, int E) {
  int e = blockIdx.x * blockDim.x + threadIdx.x;
  if (e >= E) return;
  int pos = atomicAdd(&cursor[eto[e]], 1);
  src_sorted[pos] = efrom[e];
}

// ---------------- neighbor aggregate via CSR (atomic-free), bf16 output ----------------

__global__ __launch_bounds__(256) void agg_kernel(
    const float* __restrict__ h, const int* __restrict__ src,
    const int* __restrict__ rowStart, const int* __restrict__ deg,
    unsigned short* __restrict__ outB, int n) {
  int node = blockIdx.x * 8 + (threadIdx.x >> 5);
  if (node >= n) return;
  int lane = threadIdx.x & 31;
  int c0 = lane * 4;
  int s0 = rowStart[node];
  int d = deg[node];
  float4 acc = make_float4(0.f, 0.f, 0.f, 0.f);
  for (int j = 0; j < d; ++j) {
    int u = src[s0 + j];
    float4 v = *(const float4*)(h + (size_t)u * DDIM + c0);
    acc.x += v.x; acc.y += v.y; acc.z += v.z; acc.w += v.w;
  }
  uint2 pk;
  pk.x = (unsigned)f2bf(acc.x) | ((unsigned)f2bf(acc.y) << 16);
  pk.y = (unsigned)f2bf(acc.z) | ((unsigned)f2bf(acc.w) << 16);
  *(uint2*)(outB + (size_t)node * DDIM + c0) = pk;
}

// ---------------- weight packing into WMMA B-fragment layout ----------------
// Wp per matrix: [kt(4)][nt(8)][lane(32)][i(16)] bf16
// element: k = kt*32 + (lane<16 ? 0 : 16) + i ; n = nt*16 + (lane&15)

__global__ void pack_weights_kernel(const float* __restrict__ W,
                                    unsigned short* __restrict__ Wp, int nmat) {
  int idx = blockIdx.x * blockDim.x + threadIdx.x;
  if (idx >= nmat * PACK_PER) return;
  int m = idx / PACK_PER;
  int r = idx % PACK_PER;
  int kt = r >> 12;
  int nt = (r >> 9) & 7;
  int lane = (r >> 4) & 31;
  int i = r & 15;
  int k = kt * 32 + ((lane < 16) ? 0 : 16) + i;
  int n = nt * 16 + (lane & 15);
  Wp[idx] = f2bf(W[(size_t)m * DDIM * DDIM + (size_t)k * DDIM + n]);
}

// ---------------- bf16 WMMA GEMM ----------------
// out = A @ W + bias [+ addend] [+ e2n(row,col)]
// Weights staged to 32KB LDS by the Tensor Data Mover (wave 0 issues the D#),
// 8 waves x 2 row-tiles, 64 v_wmma per wave, all unconditional (EXEC all-ones).

__device__ __forceinline__ void tdm_load_weights(const unsigned short* Wp, unsigned short* sW) {
  unsigned long long ga = (unsigned long long)Wp;
  unsigned int lds = (unsigned int)(unsigned long long)(void*)sW;  // low 32 bits = LDS byte addr
  u32x4 g0;
  g0[0] = 1u;                                             // count=1, user D#
  g0[1] = lds;                                            // lds_addr
  g0[2] = (unsigned int)(ga & 0xFFFFFFFFu);               // global_addr[31:0]
  g0[3] = ((unsigned int)(ga >> 32) & 0x01FFFFFFu) | 0x80000000u;  // addr[56:32] | type=2
  i32x8 g1;
  g1[0] = (int)0x00030000;   // data_size=3 (8B), wg_mask=0
  g1[1] = (int)0x10000000;   // tensor_dim0[15:0]=0x1000 in [63:48]
  g1[2] = (int)0x00010000;   // tensor_dim0[31:16]=0 ; tensor_dim1[15:0]=1
  g1[3] = (int)0x10000000;   // tensor_dim1[31:16]=0 ; tile_dim0=0x1000 (4096 x 8B = 32KB)
  g1[4] = (int)0x00000001;   // tile_dim1=1, tile_dim2=0
  g1[5] = 4096;              // tensor_dim0_stride[31:0]
  g1[6] = 0;
  g1[7] = 0;
  i32x4 gz = {};
#if __clang_major__ >= 23
  i32x8 gz8 = {};
  __builtin_amdgcn_tensor_load_to_lds(g0, g1, gz, gz, gz8, 0);
#else
  __builtin_amdgcn_tensor_load_to_lds(g0, g1, gz, gz, 0);
#endif
  __builtin_amdgcn_s_wait_tensorcnt(0);
}

__global__ __launch_bounds__(256) void gemm_bf16_kernel(
    const unsigned short* __restrict__ A,   // [rows][128] bf16 row-major
    const unsigned short* __restrict__ Wp,  // packed fragments (16384 bf16)
    const float* __restrict__ bias,         // [128]
    const float* __restrict__ addend,       // [rows][128] f32 or null
    const float* __restrict__ efsum,        // [rows][4] or null
    const int*   __restrict__ deg,          // [rows] (with efsum)
    const float* __restrict__ We,           // [3][128] (with efsum)
    const float* __restrict__ be,           // [128]    (with efsum)
    float* __restrict__ out,                // [rows][128]
    int rows) {
  __shared__ unsigned short sW[PACK_PER];   // 32 KB
  if (threadIdx.x < 32) tdm_load_weights(Wp, sW);  // TDM DMA, wave 0 only
  __syncthreads();

  int wave = threadIdx.x >> 5;
  int lane = threadIdx.x & 31;
  int halfsel = lane >> 4;
  int mr = lane & 15;
  int t0 = blockIdx.x * 16 + wave;
  int t1 = t0 + 8;
  bool do0 = (t0 * 16) < rows;
  bool do1 = (t1 * 16) < rows;
  if (!do0) return;
  int t1c = do1 ? t1 : t0;                  // clamp: keep WMMA path branch-free

  const unsigned short* a0 = A + (size_t)(t0 * 16 + mr) * DDIM;
  const unsigned short* a1 = A + (size_t)(t1c * 16 + mr) * DDIM;

  // all A fragments up front (one load clause; k-loop is LDS + WMMA only)
  FragU af0[4], af1[4];
#pragma unroll
  for (int kt = 0; kt < 4; ++kt) {
    int ka = kt * 32 + halfsel * 16;
    af0[kt].u[0] = *(const v4u*)(a0 + ka);
    af0[kt].u[1] = *(const v4u*)(a0 + ka + 8);
    af1[kt].u[0] = *(const v4u*)(a1 + ka);
    af1[kt].u[1] = *(const v4u*)(a1 + ka + 8);
  }

  v8f acc0[8] = {};
  v8f acc1[8] = {};
#pragma unroll
  for (int kt = 0; kt < 4; ++kt) {
    const unsigned short* wk = sW + ((size_t)kt * 256 + lane) * 16;
#pragma unroll
    for (int nt = 0; nt < 8; ++nt) {
      FragU bf;
      const v4u* bp = (const v4u*)(wk + (size_t)nt * 512);
      bf.u[0] = bp[0];
      bf.u[1] = bp[1];
      acc0[nt] = __builtin_amdgcn_wmma_f32_16x16x32_bf16(
          false, af0[kt].v, false, bf.v, (short)0, acc0[nt], false, false);
      acc1[nt] = __builtin_amdgcn_wmma_f32_16x16x32_bf16(
          false, af1[kt].v, false, bf.v, (short)0, acc1[nt], false, false);
    }
  }

#pragma unroll 1
  for (int p = 0; p < 2; ++p) {
    if (p == 1 && !do1) break;               // uniform: guards stores only
    v8f* acc = (p == 0) ? acc0 : acc1;
    int m0 = ((p == 0) ? t0 : t1) * 16;
#pragma unroll
    for (int r = 0; r < 8; ++r) {
      int row = m0 + r + halfsel * 8;
      float e0 = 0.f, e1 = 0.f, e2 = 0.f, db = 0.f;
      if (efsum) {
        e0 = efsum[(size_t)row * 4 + 0];
        e1 = efsum[(size_t)row * 4 + 1];
        e2 = efsum[(size_t)row * 4 + 2];
        db = (float)deg[row];
      }
#pragma unroll
      for (int nt = 0; nt < 8; ++nt) {
        int col = nt * 16 + mr;
        float v = acc[nt][r] + bias[col];
        if (efsum) v += e0 * We[col] + e1 * We[DDIM + col] + e2 * We[2 * DDIM + col] + db * be[col];
        if (addend) v += addend[(size_t)row * DDIM + col];
        out[(size_t)row * DDIM + col] = v;
      }
    }
  }
}

// ---------------- BN stats: relu fused; per-column sum / sumsq ----------------

__global__ __launch_bounds__(256) void bn_stats_kernel(
    const float* __restrict__ X, float* __restrict__ stats, int rows) {
  __shared__ float s_sum[256];
  __shared__ float s_sq[256];
  int d = threadIdx.x & 127;
  int half = threadIdx.x >> 7;
  int r0 = blockIdx.x * 128;
  int rend = r0 + 128; if (rend > rows) rend = rows;
  float sum = 0.f, sq = 0.f;
  for (int r = r0 + half; r < rend; r += 2) {
    float x = X[(size_t)r * DDIM + d];
    x = fmaxf(x, 0.f);
    sum += x; sq += x * x;
  }
  s_sum[threadIdx.x] = sum;
  s_sq[threadIdx.x] = sq;
  __syncthreads();
  if (half == 0) {
    atomicAdd(&stats[d], s_sum[d] + s_sum[d + 128]);
    atomicAdd(&stats[DDIM + d], s_sq[d] + s_sq[d + 128]);
  }
}

// ---------------- BN normalize (relu fused), float4; optional f32 / bf16 outputs ----------------

__global__ void bn_norm4_kernel(const float* __restrict__ X, const float* __restrict__ stats,
                                const float* __restrict__ gam, const float* __restrict__ bet,
                                float* __restrict__ outF, unsigned short* __restrict__ outB,
                                long long n4, float invN) {
  long long q = (long long)blockIdx.x * blockDim.x + threadIdx.x;
  if (q >= n4) return;
  long long i = q * 4;
  int d = (int)(i & (DDIM - 1));
  float4 x = *(const float4*)(X + i);
  float r[4] = {x.x, x.y, x.z, x.w};
  float y[4];
#pragma unroll
  for (int j = 0; j < 4; ++j) {
    int dj = d + j;
    float mean = stats[dj] * invN;
    float var = stats[DDIM + dj] * invN - mean * mean;
    float xv = fmaxf(r[j], 0.f);
    y[j] = gam[dj] * (xv - mean) * rsqrtf(var + 1e-5f) + bet[dj];
  }
  if (outF) {
    float4 o; o.x = y[0]; o.y = y[1]; o.z = y[2]; o.w = y[3];
    *(float4*)(outF + i) = o;
  }
  if (outB) {
    uint2 pk;
    pk.x = (unsigned)f2bf(y[0]) | ((unsigned)f2bf(y[1]) << 16);
    pk.y = (unsigned)f2bf(y[2]) | ((unsigned)f2bf(y[3]) << 16);
    *(uint2*)(outB + i) = pk;
  }
}

// ---------------- segment max ----------------

__global__ __launch_bounds__(128) void segmax_kernel(
    const float* __restrict__ h, const int* __restrict__ gidx,
    float* __restrict__ gemb, int rows) {
  int d = threadIdx.x;
  int r0 = blockIdx.x * 64;
  if (r0 >= rows) return;
  int rend = r0 + 64; if (rend > rows) rend = rows;
  int curg = gidx[r0];
  float m = -3.402823466e38f;
  for (int r = r0; r < rend; ++r) {
    int gi = gidx[r];
    if (gi != curg) {
      atomicMaxF(&gemb[(size_t)curg * DDIM + d], m);
      curg = gi;
      m = -3.402823466e38f;
    }
    m = fmaxf(m, h[(size_t)r * DDIM + d]);
  }
  atomicMaxF(&gemb[(size_t)curg * DDIM + d], m);
}

// ---------------- readout ----------------

__global__ __launch_bounds__(128) void readout_kernel(
    const float* __restrict__ gemb, const float* __restrict__ W,
    const float* __restrict__ b, float* __restrict__ outp) {
  int g = blockIdx.x;
  int o = threadIdx.x;
  float acc = b[o];
  const float* gr = gemb + (size_t)g * DDIM;
#pragma unroll 4
  for (int k = 0; k < DDIM; ++k) acc += gr[k] * W[(size_t)k * DDIM + o];
  outp[(size_t)g * DDIM + o] = fmaxf(acc, 0.f);
}

// ---------------- host orchestration ----------------

extern "C" void kernel_launch(void* const* d_in, const int* in_sizes, int n_in,
                              void* d_out, int out_size, void* d_ws, size_t ws_size,
                              hipStream_t stream) {
  (void)n_in; (void)out_size; (void)ws_size;
  const float* node_feat = (const float*)d_in[0];
  const float* edge_feat = (const float*)d_in[1];
  const int*   efrom     = (const int*)d_in[2];
  const int*   eto       = (const int*)d_in[3];
  const int*   gidx      = (const int*)d_in[4];
  const float* w_n2l_w   = (const float*)d_in[6];
  const float* w_n2l_b   = (const float*)d_in[7];
  const float* w_e2l_w   = (const float*)d_in[8];   // [4][3][128]
  const float* w_e2l_b   = (const float*)d_in[9];   // [4][128]
  const float* conv_w    = (const float*)d_in[10];  // [3][128][128]
  const float* conv_b    = (const float*)d_in[11];
  const float* l2_w      = (const float*)d_in[12];
  const float* l2_b      = (const float*)d_in[13];
  const float* msg_bn_g  = (const float*)d_in[14];
  const float* msg_bn_b  = (const float*)d_in[15];
  const float* hid_bn_g  = (const float*)d_in[16];
  const float* hid_bn_b  = (const float*)d_in[17];
  const float* readout_w = (const float*)d_in[18];
  const float* readout_b = (const float*)d_in[19];
  float* outp = (float*)d_out;

  const int N = in_sizes[0] / DDIM;
  const int E = in_sizes[1] / 3;
  const int G = 64;

  size_t off = 0;
  char* wsb = (char*)d_ws;
  auto take = [&](size_t bytes) {
    char* p = wsb + off;
    off += (bytes + 255) & ~(size_t)255;
    return p;
  };
  float* X            = (float*)take((size_t)N * DDIM * 4);
  float* h            = (float*)take((size_t)N * DDIM * 4);
  unsigned short* Abf = (unsigned short*)take((size_t)N * DDIM * 2);
  unsigned short* Wp  = (unsigned short*)take((size_t)7 * PACK_PER * 2);
  float* stats        = (float*)take(2 * DDIM * 4);
  float* gemb         = (float*)take((size_t)G * DDIM * 4);
  int*   deg          = (int*)take((size_t)N * 4);
  float* efsum        = (float*)take((size_t)N * 4 * 4);
  int*   rowStart     = (int*)take((size_t)N * 4);
  int*   cursor       = (int*)take((size_t)N * 4);
  int*   blockSums    = (int*)take(512 * 4);
  int*   src_sorted   = (int*)take((size_t)E * 4);

  const long long nEl = (long long)N * DDIM;
  const long long n4  = nEl / 4;
  const int v4Blocks  = (int)((n4 + 255) / 256);
  const int eBlocks   = (E + 255) / 256;
  const int nBlocks   = (N + 255) / 256;        // scan blocks (<= 512)
  const int tiles     = (N + 15) / 16;
  const int gemmBlk   = (tiles + 15) / 16;
  const int statBlk   = (N + 127) / 128;
  const float invN    = 1.0f / (float)N;

  // ---- CSR build + per-node edge-feature sums ----
  fill_kernel<<<(N + 255) / 256, 256, 0, stream>>>((float*)deg, 0.f, N);
  fill_kernel<<<(N * 4 + 255) / 256, 256, 0, stream>>>(efsum, 0.f, (long long)N * 4);
  csr_count_kernel<<<eBlocks, 256, 0, stream>>>(eto, edge_feat, deg, efsum, E);
  scan1_kernel<<<nBlocks, 256, 0, stream>>>(deg, rowStart, blockSums, N);
  scan2_kernel<<<1, 512, 0, stream>>>(blockSums, nBlocks);
  scan3_kernel<<<nBlocks, 256, 0, stream>>>(rowStart, blockSums, cursor, N);
  csr_fill_kernel<<<eBlocks, 256, 0, stream>>>(efrom, eto, cursor, src_sorted, E);

  // ---- pack weights: slot 0 = n2l, 1..3 = conv, 4..6 = l2 ----
  pack_weights_kernel<<<(1 * PACK_PER + 255) / 256, 256, 0, stream>>>(w_n2l_w, Wp, 1);
  pack_weights_kernel<<<(3 * PACK_PER + 255) / 256, 256, 0, stream>>>(conv_w, Wp + 1 * PACK_PER, 3);
  pack_weights_kernel<<<(3 * PACK_PER + 255) / 256, 256, 0, stream>>>(l2_w, Wp + 4 * PACK_PER, 3);

  // ---- stage 0 ----
  convert_bf16x4_kernel<<<v4Blocks, 256, 0, stream>>>(node_feat, Abf, n4);
  gemm_bf16_kernel<<<gemmBlk, 256, 0, stream>>>(
      Abf, Wp, w_n2l_b, (const float*)nullptr,
      efsum, deg, w_e2l_w, w_e2l_b, X, N);
  fill_kernel<<<1, 256, 0, stream>>>(stats, 0.f, 2 * DDIM);
  bn_stats_kernel<<<statBlk, 256, 0, stream>>>(X, stats, N);
  bn_norm4_kernel<<<v4Blocks, 256, 0, stream>>>(X, stats, msg_bn_g, msg_bn_b, h,
                                                (unsigned short*)nullptr, n4, invN);

  for (int lv = 0; lv < 3; ++lv) {
    agg_kernel<<<(N + 7) / 8, 256, 0, stream>>>(h, src_sorted, rowStart, deg, Abf, N);
    gemm_bf16_kernel<<<gemmBlk, 256, 0, stream>>>(
        Abf, Wp + (size_t)(1 + lv) * PACK_PER, conv_b + (size_t)lv * DDIM,
        (const float*)nullptr, efsum, deg,
        w_e2l_w + (size_t)(lv + 1) * 3 * DDIM, w_e2l_b + (size_t)(lv + 1) * DDIM, X, N);
    fill_kernel<<<1, 256, 0, stream>>>(stats, 0.f, 2 * DDIM);
    bn_stats_kernel<<<statBlk, 256, 0, stream>>>(X, stats, N);
    bn_norm4_kernel<<<v4Blocks, 256, 0, stream>>>(
        X, stats, hid_bn_g + (size_t)lv * DDIM, hid_bn_b + (size_t)lv * DDIM,
        (float*)nullptr, Abf, n4, invN);            // merged -> bf16 (next A)
    gemm_bf16_kernel<<<gemmBlk, 256, 0, stream>>>(
        Abf, Wp + (size_t)(4 + lv) * PACK_PER, l2_b + (size_t)lv * DDIM,
        h, (const float*)nullptr, (const int*)nullptr,
        (const float*)nullptr, (const float*)nullptr, X, N);
    fill_kernel<<<1, 256, 0, stream>>>(stats, 0.f, 2 * DDIM);
    bn_stats_kernel<<<statBlk, 256, 0, stream>>>(X, stats, N);
    bn_norm4_kernel<<<v4Blocks, 256, 0, stream>>>(
        X, stats, msg_bn_g + (size_t)(lv + 1) * DDIM, msg_bn_b + (size_t)(lv + 1) * DDIM,
        h, (unsigned short*)nullptr, n4, invN);
  }

  // ---- readout ----
  fill_kernel<<<(G * DDIM + 255) / 256, 256, 0, stream>>>(gemb, -3.402823466e38f, (long long)G * DDIM);
  segmax_kernel<<<(N + 63) / 64, 128, 0, stream>>>(h, gidx, gemb, N);
  readout_kernel<<<G, 128, 0, stream>>>(gemb, readout_w, readout_b, outp);
}
